// ScaledDotProductAttention_51350628991092
// MI455X (gfx1250) — matrix operand order repaired and checked by reading, not compiled, not run on hardware
//
#include <hip/hip_runtime.h>

// ---------------------------------------------------------------------------
// Flash-attention for B=4,H=16,S=2048,d=64, fp32 I/O, bool mask -> score=0.
// f16 WMMA (v_wmma_f32_16x16x32_f16) with f32 accumulation, wave32, gfx1250.
// Mask tiles staged with GLOBAL_LOAD_ASYNC_TO_LDS_B128 (ASYNCcnt path).
// ---------------------------------------------------------------------------

#define DEVINL static __device__ __forceinline__

typedef __attribute__((ext_vector_type(16))) _Float16 v16h;
typedef __attribute__((ext_vector_type(8)))  _Float16 v8h;
typedef __attribute__((ext_vector_type(4)))  _Float16 v4h;
typedef __attribute__((ext_vector_type(2)))  _Float16 v2h;
typedef __attribute__((ext_vector_type(8)))  float    v8f;
typedef __attribute__((ext_vector_type(4)))  float    v4f;

constexpr int S_LEN = 2048;
constexpr int D_DIM = 64;
constexpr int TQ    = 128;            // q rows per block
constexpr int TK    = 32;             // keys per iteration
constexpr int NW    = 8;              // waves per block
constexpr int NTH   = NW * 32;        // 256 threads
constexpr int NIT   = S_LEN / TK;     // 64 key tiles

// padded LDS row strides (in halves) -> 16B aligned rows, bank-spread
constexpr int KSTR = D_DIM + 8;       // 72 halves = 144 B
constexpr int VSTR = TK + 8;          // 40 halves = 80 B
constexpr int PSTR = D_DIM + 8;       // 72 halves (Q staging first, then P)

struct LdsLayout {
  _Float16      K[2][TK][KSTR];       // [key][d]   (B-frag source for Q*K^T)
  _Float16      Vt[2][D_DIM][VSTR];   // [d][key]   (B-frag source for P*V)
  unsigned char M[2][TQ][TK];         // mask bytes [qrow][key] (async-copied)
  _Float16      P[NW][16][PSTR];      // per-wave: Q strip at init, then P
};

DEVINL v16h cat8(v8h lo, v8h hi) {
  return __builtin_shufflevector(lo, hi, 0,1,2,3,4,5,6,7,8,9,10,11,12,13,14,15);
}

DEVINL v2h pk2(float a, float b) {
  return __builtin_bit_cast(v2h, __builtin_amdgcn_cvt_pkrtz(a, b));
}

// Flat __shared__ pointer -> LDS byte address (ISA: LDS aperture truncates to
// addr[31:0]); used as the VDST operand of async global->LDS copies.
DEVINL unsigned lds_addr(const void* p) { return (unsigned)(uintptr_t)p; }

// 16B-per-lane async copy global -> LDS, tracked by ASYNCcnt.
DEVINL void async_copy_b128(const void* gsrc, const void* ldst) {
  asm volatile("global_load_async_to_lds_b128 %0, %1, off"
               :: "v"(lds_addr(ldst)),
                  "v"((unsigned long long)(uintptr_t)gsrc)
               : "memory");
}
DEVINL void wait_async0() {
  asm volatile("s_wait_asynccnt 0x0" ::: "memory");
}

// A-matrix 16x32 f16 fragment (ISA 7.12.2): lane<16 -> row=lane, K 0..7 & 16..23
// lane>=16 -> row=lane-16, K 8..15 & 24..31. Rows contiguous in LDS.
DEVINL v16h load_frag_a(const _Float16* base, int stride, int lane) {
  const int r = lane & 15, h = lane >> 4;
  const _Float16* p = base + r * stride + h * 8;
  v8h lo = *(const v8h*)(p);
  v8h hi = *(const v8h*)(p + 16);
  return cat8(lo, hi);
}

// B-matrix 32x16 f16 fragment: lane<16 -> N=lane, K 0..7 & 16..23; lane>=16 -> N=lane-16,
// K 8..15 & 24..31 (matches A fragment). LDS stored [N][K] so K runs are contiguous.
DEVINL v16h load_frag_b(const _Float16* base, int stride, int lane) {
  const int n = lane & 15, h = lane >> 4;
  const _Float16* p = base + n * stride + h * 8;
  v8h lo = *(const v8h*)(p);
  v8h hi = *(const v8h*)(p + 16);
  return cat8(lo, hi);
}

DEVINL v8f wmma(v16h a, v16h b, v8f c) {
  return __builtin_amdgcn_wmma_f32_16x16x32_f16(false, a, false, b,
                                                (short)0, c, false, false);
}

// reductions within 16-lane halves (xor of 1,2,4,8 stays inside each half)
DEVINL float rmax16(float v) {
  v = fmaxf(v, __shfl_xor(v, 1, 32));
  v = fmaxf(v, __shfl_xor(v, 2, 32));
  v = fmaxf(v, __shfl_xor(v, 4, 32));
  v = fmaxf(v, __shfl_xor(v, 8, 32));
  return v;
}
DEVINL float rsum16(float v) {
  v += __shfl_xor(v, 1, 32);
  v += __shfl_xor(v, 2, 32);
  v += __shfl_xor(v, 4, 32);
  v += __shfl_xor(v, 8, 32);
  return v;
}

__global__ __launch_bounds__(NTH, 1)
void ScaledDotProductAttention_51350628991092_kernel(
    const float* __restrict__ Q, const float* __restrict__ K,
    const float* __restrict__ V, const unsigned char* __restrict__ Msk,
    float* __restrict__ Out)
{
  __shared__ LdsLayout lds;

  const int tid  = threadIdx.x;
  const int lane = tid & 31;
  const int wid  = tid >> 5;

  constexpr int QTILES = S_LEN / TQ;                 // 16
  const int bh    = blockIdx.x / QTILES;             // fused (b*H+h)
  const int qt    = blockIdx.x % QTILES;
  const int qbase = qt * TQ;

  const float* Qb = Q + (size_t)bh * S_LEN * D_DIM;
  const float* Kb = K + (size_t)bh * S_LEN * D_DIM;
  const float* Vb = V + (size_t)bh * S_LEN * D_DIM;
  const unsigned char* Mb = Msk + (size_t)bh * S_LEN * S_LEN + (size_t)qbase * S_LEN;
  float* Ob = Out + (size_t)bh * S_LEN * D_DIM;

  const int rowbase = wid * 16;      // this wave's q-strip inside the block tile
  const int n    = lane & 15;        // C-fragment column
  const int rofs = (lane >> 4) * 8;  // C-fragment row base (0 or 8)
  const float scale = 0.125f;        // 1/sqrt(64)

  // ---- stage this wave's 16x64 Q strip (f32 -> f16) into its private region
  {
    const float* qsrc = Qb + (size_t)(qbase + rowbase) * D_DIM;
#pragma unroll
    for (int k = 0; k < 8; ++k) {
      const int i   = lane + k * 32;       // 256 float4 units = 16 rows x 16
      const int row = i >> 4;
      const int c4  = i & 15;
      v4f f = *(const v4f*)(qsrc + row * D_DIM + c4 * 4);
      v2h h0 = pk2(f[0], f[1]);
      v2h h1 = pk2(f[2], f[3]);
      v4h h = { h0[0], h0[1], h1[0], h1[1] };
      *(v4h*)(&lds.P[wid][row][c4 * 4]) = h;
    }
  }
  // same-wave LDS dependency -> compiler inserts s_wait_dscnt; no barrier needed
  const v16h aq0 = load_frag_a(&lds.P[wid][0][0],  PSTR, lane);  // d 0..31
  const v16h aq1 = load_frag_a(&lds.P[wid][0][32], PSTR, lane);  // d 32..63

  // ---- staging of one 32-key tile: K f16 + V^T f16 via registers (need the
  //      f32->f16 convert), mask bytes via async global->LDS DMA (no convert).
  auto stage = [&](int it, int buf) {
    const int kb = it * TK;
    {
      const int key = tid >> 3;            // 0..31
      const int d0  = (tid & 7) * 8;       // 0..56
      const float* src = Kb + (size_t)(kb + key) * D_DIM + d0;
      v4f a = *(const v4f*)(src);
      v4f b = *(const v4f*)(src + 4);
      v2h h0 = pk2(a[0], a[1]);
      v2h h1 = pk2(a[2], a[3]);
      v2h h2 = pk2(b[0], b[1]);
      v2h h3 = pk2(b[2], b[3]);
      v8h h = { h0[0], h0[1], h1[0], h1[1], h2[0], h2[1], h3[0], h3[1] };
      *(v8h*)(&lds.K[buf][key][d0]) = h;

      const float* vs = Vb + (size_t)(kb + key) * D_DIM + d0;
      v4f va = *(const v4f*)(vs);
      v4f vb = *(const v4f*)(vs + 4);
#pragma unroll
      for (int j = 0; j < 8; ++j)
        lds.Vt[buf][d0 + j][key] = (_Float16)(j < 4 ? va[j] : vb[j - 4]);
    }
    {
      const int row = tid >> 1;            // 0..127
      const int c   = (tid & 1) * 16;
      async_copy_b128(Mb + (size_t)row * S_LEN + kb + c,
                      &lds.M[buf][row][c]);
    }
  };

  // ---- accumulators: O (16x64 f32 in 4 C-frags), running max, per-lane sum
  v8f o0 = {}, o1 = {}, o2 = {}, o3 = {};
  float mrow[8], lrow[8];
#pragma unroll
  for (int r = 0; r < 8; ++r) { mrow[r] = -1e30f; lrow[r] = 0.0f; }

  stage(0, 0);
  wait_async0();
  __syncthreads();

  for (int it = 0; it < NIT; ++it) {
    const int buf = it & 1;
    if (it + 1 < NIT) stage(it + 1, buf ^ 1);   // prefetch into other buffer

    // ---- S = Q * K^T for this 32-key tile (4 WMMAs)
    v8f s0 = {}, s1 = {};
    {
      v16h b00 = load_frag_b(&lds.K[buf][0][0],   KSTR, lane); // keys 0-15,  d 0-31
      v16h b01 = load_frag_b(&lds.K[buf][0][32],  KSTR, lane); // keys 0-15,  d 32-63
      v16h b10 = load_frag_b(&lds.K[buf][16][0],  KSTR, lane); // keys 16-31, d 0-31
      v16h b11 = load_frag_b(&lds.K[buf][16][32], KSTR, lane); // keys 16-31, d 32-63
      s0 = wmma(aq0, b00, s0);  s0 = wmma(aq1, b01, s0);
      s1 = wmma(aq0, b10, s1);  s1 = wmma(aq1, b11, s1);
    }

    // ---- scale, mask(->0), online softmax bookkeeping
    float p0[8], p1[8], alpha[8];
#pragma unroll
    for (int r = 0; r < 8; ++r) {
      const int row = rofs + r;                          // row within 16-strip
      float v0 = s0[r] * scale;
      float v1 = s1[r] * scale;
      if (lds.M[buf][rowbase + row][n])      v0 = 0.0f;  // masked_fill(mask, 0)
      if (lds.M[buf][rowbase + row][16 + n]) v1 = 0.0f;
      const float mx   = rmax16(fmaxf(v0, v1));          // uniform per row-half
      const float mnew = fmaxf(mrow[r], mx);
      const float a    = __expf(mrow[r] - mnew);
      alpha[r] = a;
      p0[r] = __expf(v0 - mnew);
      p1[r] = __expf(v1 - mnew);
      lrow[r] = lrow[r] * a + p0[r] + p1[r];             // per-lane partial sum
      mrow[r] = mnew;
    }
#pragma unroll
    for (int r = 0; r < 8; ++r) {
      o0[r] *= alpha[r]; o1[r] *= alpha[r]; o2[r] *= alpha[r]; o3[r] *= alpha[r];
    }

    // ---- C-layout P -> A-layout via per-wave LDS scratch (same-wave dep)
#pragma unroll
    for (int r = 0; r < 8; ++r) {
      const int row = rofs + r;
      lds.P[wid][row][n]      = (_Float16)p0[r];
      lds.P[wid][row][16 + n] = (_Float16)p1[r];
    }
    const v16h ap = load_frag_a(&lds.P[wid][0][0], PSTR, lane);

    // ---- O += P * V  (4 WMMAs over the 64 d-channels)
    {
      v16h bv0 = load_frag_b(&lds.Vt[buf][0][0],  VSTR, lane);
      v16h bv1 = load_frag_b(&lds.Vt[buf][16][0], VSTR, lane);
      v16h bv2 = load_frag_b(&lds.Vt[buf][32][0], VSTR, lane);
      v16h bv3 = load_frag_b(&lds.Vt[buf][48][0], VSTR, lane);
      o0 = wmma(ap, bv0, o0);
      o1 = wmma(ap, bv1, o1);
      o2 = wmma(ap, bv2, o2);
      o3 = wmma(ap, bv3, o3);
    }

    wait_async0();     // this wave's async mask copies into buf^1 landed
    __syncthreads();   // staging of buf^1 done; buf free for it+2
  }

  // ---- finalize: reduce row sums across lanes once, normalize, store fp32
#pragma unroll
  for (int r = 0; r < 8; ++r) {
    const float lsum = rsum16(lrow[r]);
    const float inv  = 1.0f / lsum;
    o0[r] *= inv; o1[r] *= inv; o2[r] *= inv; o3[r] *= inv;
  }
  float* obase = Ob + (size_t)(qbase + rowbase) * D_DIM;
#pragma unroll
  for (int r = 0; r < 8; ++r) {
    const int row = rofs + r;
    obase[row * D_DIM +      n] = o0[r];
    obase[row * D_DIM + 16 + n] = o1[r];
    obase[row * D_DIM + 32 + n] = o2[r];
    obase[row * D_DIM + 48 + n] = o3[r];
  }
}

extern "C" void kernel_launch(void* const* d_in, const int* in_sizes, int n_in,
                              void* d_out, int out_size, void* d_ws, size_t ws_size,
                              hipStream_t stream) {
  const float* Q = (const float*)d_in[0];
  const float* K = (const float*)d_in[1];
  const float* V = (const float*)d_in[2];
  const unsigned char* M = (const unsigned char*)d_in[3];  // jnp.bool_ = 1 byte
  float* Out = (float*)d_out;

  constexpr int B = 4, H = 16;
  dim3 grid(B * H * (S_LEN / TQ));   // 1024 blocks
  ScaledDotProductAttention_51350628991092_kernel<<<grid, NTH, 0, stream>>>(
      Q, K, V, M, Out);
}
